// BasicBlockR_30305289240890
// MI455X (gfx1250) — 8-bit matrix operand order repaired and checked by reading, compiled, not run on hardware
//
#include <hip/hip_runtime.h>
#include <stdint.h>

typedef __attribute__((ext_vector_type(8))) int v8i;
typedef __attribute__((ext_vector_type(4))) int v4i;

#define EPS_F 1e-5f
#define Bsz 64
#define Csz 256
#define NPIX 65536
#define KTOT 2304
#define KCHUNKS 36
#define NTILES 16
#define WELEMS 589824
#define WQ_DWORDS 147456

__global__ __launch_bounds__(256) void stats_kernel(const float* __restrict__ w1,
                                                    const float* __restrict__ w2,
                                                    float* __restrict__ stats) {
    const int conv = blockIdx.x;
    const float* w = conv ? w2 : w1;
    __shared__ float red[256];
    __shared__ float red2[256];
    __shared__ float sdelta;
    const int tid = threadIdx.x;

    float s = 0.f;
    for (int i = tid; i < WELEMS; i += 256) s += fabsf(w[i]);
    red[tid] = s; __syncthreads();
    for (int o = 128; o > 0; o >>= 1) {
        if (tid < o) red[tid] += red[tid + o];
        __syncthreads();
    }
    if (tid == 0) {
        sdelta = 0.7f * red[0] / (float)WELEMS;
        stats[2 * conv] = sdelta;
    }
    __syncthreads();
    const float delta = sdelta;

    float sa = 0.f, sc = 0.f;
    for (int i = tid; i < WELEMS; i += 256) {
        float a = fabsf(w[i]);
        if (a > delta) { sa += a; sc += 1.f; }
    }
    red[tid] = sa; red2[tid] = sc; __syncthreads();
    for (int o = 128; o > 0; o >>= 1) {
        if (tid < o) { red[tid] += red[tid + o]; red2[tid] += red2[tid + o]; }
        __syncthreads();
    }
    if (tid == 0) stats[2 * conv + 1] = red[0] / fmaxf(red2[0], 1.f);
}

__global__ __launch_bounds__(256) void quant_kernel(const float* __restrict__ w1,
                                                    const float* __restrict__ w2,
                                                    const float* __restrict__ stats,
                                                    int* __restrict__ wq1,
                                                    int* __restrict__ wq2) {
    const int idx = blockIdx.x * 256 + threadIdx.x;
    if (idx >= 2 * WQ_DWORDS) return;
    const int conv = idx / WQ_DWORDS;
    const int r = idx - conv * WQ_DWORDS;
    const float* w = conv ? w2 : w1;
    int* wq = conv ? wq2 : wq1;
    const float delta = stats[2 * conv];

    const int lane = r & 31;
    const int j    = (r >> 5) & 7;
    const int nt   = (r >> 8) & 15;
    const int kc   = r >> 12;
    const int hf   = lane >> 4;
    const int co   = nt * 16 + (lane & 15);

    uint32_t pk = 0;
#pragma unroll
    for (int b = 0; b < 4; ++b) {
        const int kl = ((j >> 2) * 32) + hf * 16 + (j & 3) * 4 + b;
        const int k  = kc * 64 + kl;
        const int ci = k & 255;
        const int rs = k >> 8;
        const float wv = w[(co * 256 + ci) * 9 + rs];
        int q = (fabsf(wv) > delta) ? (wv > 0.f ? 1 : -1) : 0;
        pk |= ((uint32_t)(uint8_t)(int8_t)q) << (8 * b);
    }
    wq[r] = pk;
}

__global__ __launch_bounds__(256) void bin1_kernel(const float* __restrict__ x,
                                                   const float* __restrict__ g,
                                                   const float* __restrict__ be,
                                                   const float* __restrict__ mu,
                                                   const float* __restrict__ va,
                                                   int8_t* __restrict__ a1) {
    const int idx = blockIdx.x * 256 + threadIdx.x;
    const int c  = (idx >> 10) & 255;
    const int b  = idx >> 18;
    const int hw = idx & 1023;
    const float inv = g[c] * rsqrtf(va[c] + EPS_F);
    const float y = x[idx] * inv + (be[c] - mu[c] * inv);
    a1[((b << 10) | hw) * 256 + c] = (y >= 0.f) ? (int8_t)1 : (int8_t)-1;
}

template <int MODE>
__global__ __launch_bounds__(256) void tbn_gemm_kernel(
    const int8_t* __restrict__ act, const int* __restrict__ wq,
    const float* __restrict__ stats, int conv,
    const float* __restrict__ g2, const float* __restrict__ b2,
    const float* __restrict__ m2, const float* __restrict__ v2,
    const float* __restrict__ xin, float* __restrict__ fout,
    int8_t* __restrict__ aout) {
    __shared__ char Alds[2][128 * 80];

    const int tid  = threadIdx.x;
    const int lane = tid & 31;
    const int wv   = tid >> 5;
    const int hf   = lane >> 4;
    const int n0   = lane & 15;
    const int blockM0 = blockIdx.x * 128;

    v8i acc[8][2];
#pragma unroll
    for (int mt = 0; mt < 8; ++mt)
#pragma unroll
        for (int q = 0; q < 2; ++q)
#pragma unroll
            for (int d = 0; d < 8; ++d) acc[mt][q][d] = 0;

    const int p    = tid >> 1;
    const int half = tid & 1;
    const int pix  = blockM0 + p;
    const int pb   = pix >> 10;
    const int ph   = (pix >> 5) & 31;
    const int pw   = pix & 31;

    auto issueA = [&](int kc, int buf) {
        const int rs    = kc >> 2;
        const int cbase = (kc & 3) << 6;
        const int rr = rs / 3, ss = rs - rr * 3;
        const int hy = ph + rr - 1;
        const int wx = pw + ss - 1;
        char* dst = &Alds[buf][p * 80 + half * 32];
        if (hy >= 0 && hy < 32 && wx >= 0 && wx < 32) {
            const int8_t* src = act + (((pb * 32 + hy) * 32 + wx) << 8)
                                + cbase + half * 32;
            const uint32_t ldso = (uint32_t)(uintptr_t)dst;
            asm volatile(
                "global_load_async_to_lds_b128 %0, %1, off\n\t"
                "global_load_async_to_lds_b128 %0, %1, off offset:16"
                :: "v"(ldso), "v"(src) : "memory");
        } else {
            v4i z = {0, 0, 0, 0};
            v4i* d4 = (v4i*)dst;
            d4[0] = z; d4[1] = z;
        }
    };

    issueA(0, 0);
    asm volatile("s_wait_asynccnt 0x0" ::: "memory");
    __syncthreads();

    for (int kc = 0; kc < KCHUNKS; ++kc) {
        const int cur = kc & 1;
        if (kc + 1 < KCHUNKS) {
            issueA(kc + 1, cur ^ 1);
            __builtin_prefetch(wq + (((kc + 1) * 16 + wv * 2) * 8) * 32, 0, 1);
        }

        v8i bf[2];
#pragma unroll
        for (int q = 0; q < 2; ++q) {
            const int nt = wv * 2 + q;
            const int* bp = wq + ((kc * 16 + nt) * 8) * 32 + lane;
#pragma unroll
            for (int j = 0; j < 8; ++j) bf[q][j] = bp[j * 32];
        }

#pragma unroll
        for (int mt = 0; mt < 8; ++mt) {
            const uint32_t* ar = (const uint32_t*)(&Alds[cur][(mt * 16 + n0) * 80]);
            v8i af;
#pragma unroll
            for (int j = 0; j < 8; ++j) af[j] = ar[(j >> 2) * 8 + hf * 4 + (j & 3)];
            acc[mt][0] = __builtin_amdgcn_wmma_i32_16x16x64_iu8(
                true, af, true, bf[0], acc[mt][0], false, false);
            acc[mt][1] = __builtin_amdgcn_wmma_i32_16x16x64_iu8(
                true, af, true, bf[1], acc[mt][1], false, false);
        }

        asm volatile("s_wait_asynccnt 0x0" ::: "memory");
        asm volatile("s_wait_dscnt 0x0" ::: "memory");
        __syncthreads();
    }

    const float alpha = stats[2 * conv + 1];
#pragma unroll
    for (int q = 0; q < 2; ++q) {
        const int co = (wv * 2 + q) * 16 + n0;
        float sc = 0.f, bi = 0.f;
        if (MODE == 0) {
            const float inv = g2[co] * rsqrtf(v2[co] + EPS_F);
            sc = inv; bi = b2[co] - m2[co] * inv;
        }
#pragma unroll
        for (int mt = 0; mt < 8; ++mt) {
#pragma unroll
            for (int d = 0; d < 8; ++d) {
                const int pixe = blockM0 + mt * 16 + d + 8 * hf;
                const float val = alpha * (float)acc[mt][q][d];
                if (MODE == 0) {
                    const float y = val * sc + bi;
                    aout[pixe * 256 + co] = (y >= 0.f) ? (int8_t)1 : (int8_t)-1;
                } else {
                    const int b  = pixe >> 10;
                    const int hw = pixe & 1023;
                    const int oidx = (b * 256 + co) * 1024 + hw;
                    fout[oidx] = xin[oidx] + val;
                }
            }
        }
    }
}

extern "C" void kernel_launch(void* const* d_in, const int* in_sizes, int n_in,
                              void* d_out, int out_size, void* d_ws, size_t ws_size,
                              hipStream_t stream) {
    const float* x   = (const float*)d_in[0];
    const float* g1  = (const float*)d_in[1];
    const float* be1 = (const float*)d_in[2];
    const float* mu1 = (const float*)d_in[3];
    const float* va1 = (const float*)d_in[4];
    const float* w1  = (const float*)d_in[5];
    const float* g2  = (const float*)d_in[6];
    const float* be2 = (const float*)d_in[7];
    const float* mu2 = (const float*)d_in[8];
    const float* va2 = (const float*)d_in[9];
    const float* w2  = (const float*)d_in[10];
    float* out = (float*)d_out;

    char* ws = (char*)d_ws;
    float* stats = (float*)ws;
    int*    wq1  = (int*)(ws + 256);
    int*    wq2  = (int*)(ws + 256 + 589824);
    int8_t* a1   = (int8_t*)(ws + 1180160);
    int8_t* a2   = (int8_t*)(ws + 1180160 + 16777216);
    (void)in_sizes; (void)n_in; (void)out_size; (void)ws_size;

    stats_kernel<<<2, 256, 0, stream>>>(w1, w2, stats);
    quant_kernel<<<(2 * WQ_DWORDS + 255) / 256, 256, 0, stream>>>(w1, w2, stats, wq1, wq2);
    bin1_kernel<<<NPIX * Csz / 256, 256, 0, stream>>>(x, g1, be1, mu1, va1, a1);
    tbn_gemm_kernel<0><<<NPIX / 128, 256, 0, stream>>>(
        a1, wq1, stats, 0, g2, be2, mu2, va2, nullptr, nullptr, a2);
    tbn_gemm_kernel<1><<<NPIX / 128, 256, 0, stream>>>(
        a2, wq2, stats, 1, nullptr, nullptr, nullptr, nullptr, x, out, nullptr);
}
